// VisionMambaDecoder_88510686036387
// MI455X (gfx1250) — matrix operand order repaired and checked by reading, not compiled, not run on hardware
//
#include <hip/hip_runtime.h>
#include <hip/hip_bf16.h>
#include <math.h>

typedef __attribute__((ext_vector_type(16))) _Float16 v16h;
typedef __attribute__((ext_vector_type(8)))  float    v8f;

#define DSTATE 8

__device__ __forceinline__ float sigf(float x) { return 1.f / (1.f + __expf(-x)); }

// ---------------------------------------------------------------------------
// Generic skinny GEMM: Cout(MxN) = A(MxK) @ W(NxK)^T [+bias][+gelu]
// One wave32 per 16x16 output tile, f16 WMMA with f32 accumulate.
// K is compile-time (16/32/64/128) so all guards constant-fold and the
// k-loop unrolls into straight-line b128 loads + v_wmma.
// ---------------------------------------------------------------------------
template <int K, int ACT>
__global__ __launch_bounds__(128)
void vmd_gemm_wmma(const float* __restrict__ A, const float* __restrict__ W,
                   const float* __restrict__ bias, float* __restrict__ Cout,
                   int M, int N)
{
    const int wave = threadIdx.x >> 5;
    const int lane = threadIdx.x & 31;
    const int ntiles = N >> 4;
    const int tile = blockIdx.x * 4 + wave;
    const int mt = tile / ntiles;
    const int nt = tile - mt * ntiles;
    const int m0 = mt << 4;
    if (m0 >= M) return;
    const int n0 = nt << 4;
    const int half = lane >> 4;       // K-subgroup / M-subgroup selector
    const int mr   = lane & 15;

    const float* __restrict__ Arow = A + (size_t)(m0 + mr) * K;
    __builtin_prefetch(Arow, 0, 1);   // global_prefetch_b8

    v8f acc = {};
#pragma unroll
    for (int k0 = 0; k0 < K; k0 += 32) {
        // ---- A: 16x32 f16 tile. a[8g+e] <-> K = k0 + 16g + 8*half + e ----
        v16h a;
#pragma unroll
        for (int g = 0; g < 2; ++g) {
            if (k0 + g * 16 < K) {   // compile-time uniform (K multiple of 16)
                const int kb = k0 + g * 16 + half * 8;
                float4 x0 = *(const float4*)(Arow + kb);
                float4 x1 = *(const float4*)(Arow + kb + 4);
                a[8 * g + 0] = (_Float16)x0.x;
                a[8 * g + 1] = (_Float16)x0.y;
                a[8 * g + 2] = (_Float16)x0.z;
                a[8 * g + 3] = (_Float16)x0.w;
                a[8 * g + 4] = (_Float16)x1.x;
                a[8 * g + 5] = (_Float16)x1.y;
                a[8 * g + 6] = (_Float16)x1.z;
                a[8 * g + 7] = (_Float16)x1.w;
            } else {
#pragma unroll
                for (int e = 0; e < 8; ++e) a[8 * g + e] = (_Float16)0.f;
            }
        }
        // ---- B: 32x16 f16 tile. lane = N col; b[8g+e] <-> K = k0 + 16g + 8*half + e; B[k][n] = W[n][k] ----
        v16h b = {};
        const float* __restrict__ Wrow = W + (size_t)(n0 + mr) * K;
#pragma unroll
        for (int g = 0; g < 2; ++g) {
            if (k0 + g * 16 < K) {
                const int kb = k0 + g * 16 + half * 8;
#pragma unroll
                for (int e = 0; e < 8; ++e) b[8 * g + e] = (_Float16)Wrow[kb + e];
            }
        }
        acc = __builtin_amdgcn_wmma_f32_16x16x32_f16(false, a, false, b,
                                                     (short)0, acc, false, false);
    }
    // D layout: VGPR r, lane l -> M = m0 + r + 8*(l>>4), N = n0 + (l&15)
    const int nn = n0 + (lane & 15);
    float bv = bias ? bias[nn] : 0.f;
#pragma unroll
    for (int r = 0; r < 8; ++r) {
        int m = m0 + r + 8 * half;
        float v = acc[r] + bv;
        if (ACT == 1) v = 0.5f * v * (1.f + erff(v * 0.70710678f)); // exact gelu
        Cout[(size_t)m * N + nn] = v;
    }
}

// ---------------------------------------------------------------------------
// ConvTranspose2d k=4 s=2 p=1.  Input either NCHW (flag 0) or (L,C) (flag 1).
// Output channel-major (Cout, 2H, 2W).
// ---------------------------------------------------------------------------
__global__ void vmd_convt4x4(const float* __restrict__ in, int in_is_lc,
                             const float* __restrict__ w, const float* __restrict__ bias,
                             float* __restrict__ out, int Cin, int Cout, int Hin, int Win)
{
    int Hout = 2 * Hin, Wout = 2 * Win;
    int idx = blockIdx.x * blockDim.x + threadIdx.x;
    int total = Cout * Hout * Wout;
    if (idx >= total) return;
    int ox = idx % Wout;
    int oy = (idx / Wout) % Hout;
    int oc = idx / (Wout * Hout);
    float acc = bias[oc];
    for (int ky = 0; ky < 4; ++ky) {
        int t = oy - 2 + ky;
        if (t < 0 || (t & 1)) continue;
        int iy = t >> 1; if (iy >= Hin) continue;
        for (int kx = 0; kx < 4; ++kx) {
            int u = ox - 2 + kx;
            if (u < 0 || (u & 1)) continue;
            int ix = u >> 1; if (ix >= Win) continue;
            const float* wp = w + ((size_t)(3 - ky) * 4 + (3 - kx)) + (size_t)oc * 16;
            for (int ic = 0; ic < Cin; ++ic) {
                float xv = in_is_lc ? in[(size_t)(iy * Win + ix) * Cin + ic]
                                    : in[((size_t)ic * Hin + iy) * Win + ix];
                acc += xv * wp[(size_t)ic * Cout * 16];
            }
        }
    }
    out[idx] = acc;
}

// InstanceNorm + LeakyReLU(0.2), in place on (C, P) buffer; one block per channel.
__global__ void vmd_inorm_lrelu(float* __restrict__ x, const float* __restrict__ g,
                                const float* __restrict__ b, int P)
{
    __shared__ float s1[256], s2[256];
    int c = blockIdx.x;
    float* xc = x + (size_t)c * P;
    float s = 0.f, q = 0.f;
    for (int i = threadIdx.x; i < P; i += 256) { float v = xc[i]; s += v; q += v * v; }
    s1[threadIdx.x] = s; s2[threadIdx.x] = q;
    __syncthreads();
    for (int o = 128; o; o >>= 1) {
        if (threadIdx.x < o) { s1[threadIdx.x] += s1[threadIdx.x + o]; s2[threadIdx.x] += s2[threadIdx.x + o]; }
        __syncthreads();
    }
    float mean = s1[0] / P, var = s2[0] / P - mean * mean;
    float sc = g[c] * rsqrtf(var + 1e-5f), sh = b[c] - mean * sc;
    for (int i = threadIdx.x; i < P; i += 256) {
        float v = xc[i] * sc + sh;
        xc[i] = v > 0.f ? v : 0.2f * v;
    }
}

// 1x1 fuse conv over concat([up, skip]) (both (C,P)); output transposed to (P, C).
__global__ void vmd_fuse1x1(const float* __restrict__ up, const float* __restrict__ skip,
                            const float* __restrict__ w, const float* __restrict__ b,
                            float* __restrict__ t, int C, int P)
{
    int idx = blockIdx.x * blockDim.x + threadIdx.x;
    if (idx >= P * C) return;
    int oc = idx % C;
    int p = idx / C;
    const float* wr = w + (size_t)oc * 2 * C;
    float acc = b[oc];
    for (int ic = 0; ic < C; ++ic) acc += wr[ic] * up[(size_t)ic * P + p];
    for (int ic = 0; ic < C; ++ic) acc += wr[C + ic] * skip[(size_t)ic * P + p];
    t[idx] = acc;
}

// LayerNorm over last dim C (<=32), in place; one wave per row.
__global__ void vmd_layernorm(float* __restrict__ t, const float* __restrict__ g,
                              const float* __restrict__ b, int L, int C)
{
    int wave = threadIdx.x >> 5, lane = threadIdx.x & 31;
    int row = blockIdx.x * 8 + wave;
    if (row >= L) return;
    float v = (lane < C) ? t[(size_t)row * C + lane] : 0.f;
    float s = v, q = v * v;
    for (int o = 16; o; o >>= 1) { s += __shfl_xor(s, o, 32); q += __shfl_xor(q, o, 32); }
    float mean = s / C, var = q / C - mean * mean;
    if (lane < C)
        t[(size_t)row * C + lane] = (v - mean) * rsqrtf(var + 1e-5f) * g[lane] + b[lane];
}

// Depthwise causal conv1d k=3 over sequence + SiLU. xz row stride = 2*Di.
__global__ void vmd_dwconv_silu(const float* __restrict__ xz, const float* __restrict__ w,
                                const float* __restrict__ b, float* __restrict__ xc,
                                int L, int Di)
{
    int idx = blockIdx.x * blockDim.x + threadIdx.x;
    if (idx >= L * Di) return;
    int d = idx % Di;
    int l = idx / Di;
    int ld = 2 * Di;
    float acc = b[d];
#pragma unroll
    for (int j = 0; j < 3; ++j) {
        int ll = l - 2 + j;
        if (ll >= 0) acc += w[d * 3 + j] * xz[(size_t)ll * ld + d];
    }
    xc[idx] = acc * sigf(acc);
}

// x_proj: dbl(L,Nx) = xc(L,Di) @ xw(Nx,Di)^T   (Nx = R + 16, tiny)
__global__ void vmd_xproj(const float* __restrict__ xc, const float* __restrict__ xw,
                          float* __restrict__ dbl, int L, int Di, int Nx)
{
    int idx = blockIdx.x * blockDim.x + threadIdx.x;
    if (idx >= L * Nx) return;
    int n = idx % Nx;
    int l = idx / Nx;
    const float* wr = xw + (size_t)n * Di;
    const float* xr = xc + (size_t)l * Di;
    float acc = 0.f;
    for (int k = 0; k < Di; ++k) acc += wr[k] * xr[k];
    dbl[idx] = acc;
}

// dt = softplus(dbl[:, :R] @ dt_w^T + dt_b)
__global__ void vmd_dtcalc(const float* __restrict__ dbl, const float* __restrict__ dtw,
                           const float* __restrict__ dtb, float* __restrict__ dt,
                           int L, int Di, int R, int Nx)
{
    int idx = blockIdx.x * blockDim.x + threadIdx.x;
    if (idx >= L * Di) return;
    int d = idx % Di;
    int l = idx / Di;
    float acc = dtb[d];
    for (int r = 0; r < R; ++r) acc += dtw[d * R + r] * dbl[(size_t)l * Nx + r];
    dt[idx] = (acc > 20.f) ? acc : log1pf(__expf(acc));
}

// Chunked selective-scan pass 1: per (d,s,chunk) compute (prodA, h_end).
__global__ void vmd_scan_p1(const float* __restrict__ dt, const float* __restrict__ dbl,
                            const float* __restrict__ xc, const float* __restrict__ A_log,
                            float* __restrict__ cA, float* __restrict__ cH,
                            int Di, int Nx, int R, int nch, int CL)
{
    int g = blockIdx.x * blockDim.x + threadIdx.x;
    int total = Di * DSTATE * nch;
    if (g >= total) return;
    int s = g & 7;
    int rest = g >> 3;
    int chunk = rest % nch;
    int d = rest / nch;
    float Av = -__expf(A_log[d * DSTATE + s]);
    float h = 0.f, ap = 1.f;
    int l0 = chunk * CL;
    for (int i = 0; i < CL; ++i) {
        int l = l0 + i;
        float dv = dt[(size_t)l * Di + d];
        float a = __expf(dv * Av);
        h = a * h + dv * dbl[(size_t)l * Nx + R + s] * xc[(size_t)l * Di + d];
        ap *= a;
    }
    int o = (d * DSTATE + s) * nch + chunk;
    cA[o] = ap;
    cH[o] = h;
}

// Pass 2: sequential prefix over chunks per (d,s) -> initial state per chunk.
__global__ void vmd_scan_p2(const float* __restrict__ cA, const float* __restrict__ cH,
                            float* __restrict__ cI, int Di, int nch)
{
    int g = blockIdx.x * blockDim.x + threadIdx.x;
    if (g >= Di * DSTATE) return;
    float h = 0.f;
    for (int c = 0; c < nch; ++c) {
        cI[(size_t)g * nch + c] = h;
        h = cA[(size_t)g * nch + c] * h + cH[(size_t)g * nch + c];
    }
}

// Pass 3: replay with correct h0; 8-lane shuffle reduce over state dim, write y.
__global__ void vmd_scan_p3(const float* __restrict__ dt, const float* __restrict__ dbl,
                            const float* __restrict__ xc, const float* __restrict__ A_log,
                            const float* __restrict__ cI, float* __restrict__ ys,
                            int Di, int Nx, int R, int nch, int CL)
{
    int g = blockIdx.x * blockDim.x + threadIdx.x;
    int total = Di * DSTATE * nch;
    if (g >= total) return;
    int s = g & 7;
    int rest = g >> 3;
    int chunk = rest % nch;
    int d = rest / nch;
    float Av = -__expf(A_log[d * DSTATE + s]);
    float h = cI[(size_t)(d * DSTATE + s) * nch + chunk];
    int l0 = chunk * CL;
    for (int i = 0; i < CL; ++i) {
        int l = l0 + i;
        float dv = dt[(size_t)l * Di + d];
        float a = __expf(dv * Av);
        h = a * h + dv * dbl[(size_t)l * Nx + R + s] * xc[(size_t)l * Di + d];
        float contrib = h * dbl[(size_t)l * Nx + R + DSTATE + s];
        contrib += __shfl_xor(contrib, 1, 32);
        contrib += __shfl_xor(contrib, 2, 32);
        contrib += __shfl_xor(contrib, 4, 32);
        if (s == 0) ys[(size_t)l * Di + d] = contrib;
    }
}

// y = (ys + D*xc) * silu(z)
__global__ void vmd_gate(const float* __restrict__ ys, const float* __restrict__ xc,
                         const float* __restrict__ xz, const float* __restrict__ Dp,
                         float* __restrict__ gb, int L, int Di)
{
    int idx = blockIdx.x * blockDim.x + threadIdx.x;
    if (idx >= L * Di) return;
    int d = idx % Di;
    int l = idx / Di;
    float y = ys[idx] + Dp[d] * xc[idx];
    float z = xz[(size_t)l * 2 * Di + Di + d];
    gb[idx] = y * z * sigf(z);
}

// Final 3x3 conv (C=16 -> 1) + sigmoid from (H*W, C) layout.
__global__ void vmd_outconv(const float* __restrict__ t, const float* __restrict__ w,
                            const float* __restrict__ b, float* __restrict__ out,
                            int C, int H, int Wd)
{
    int idx = blockIdx.x * blockDim.x + threadIdx.x;
    if (idx >= H * Wd) return;
    int x = idx % Wd;
    int y = idx / Wd;
    float acc = b[0];
    for (int ky = 0; ky < 3; ++ky) {
        int yy = y + ky - 1;
        if (yy < 0 || yy >= H) continue;
        for (int kx = 0; kx < 3; ++kx) {
            int xx = x + kx - 1;
            if (xx < 0 || xx >= Wd) continue;
            const float* tr = t + (size_t)(yy * Wd + xx) * C;
            for (int ic = 0; ic < C; ++ic) acc += tr[ic] * w[(ic * 3 + ky) * 3 + kx];
        }
    }
    out[idx] = sigf(acc);
}

// ---------------------------------------------------------------------------
static void launch_gemm(const float* A, const float* W, const float* bias, float* C,
                        int M, int K, int N, int act, hipStream_t s)
{
    int tiles = (M / 16) * (N / 16);
    int blocks = (tiles + 3) / 4;
#define VMD_GO(KK)                                                                  \
    do {                                                                            \
        if (act == 1)                                                               \
            vmd_gemm_wmma<KK, 1><<<blocks, 128, 0, s>>>(A, W, bias, C, M, N);       \
        else                                                                        \
            vmd_gemm_wmma<KK, 0><<<blocks, 128, 0, s>>>(A, W, bias, C, M, N);       \
    } while (0)
    switch (K) {
        case 16:  VMD_GO(16);  break;
        case 32:  VMD_GO(32);  break;
        case 64:  VMD_GO(64);  break;
        case 128: VMD_GO(128); break;
        default: break;
    }
#undef VMD_GO
}

extern "C" void kernel_launch(void* const* d_in, const int* in_sizes, int n_in,
                              void* d_out, int out_size, void* d_ws, size_t ws_size,
                              hipStream_t stream)
{
    (void)in_sizes; (void)n_in; (void)out_size; (void)ws_size;
    auto P = [&](int i) -> const float* { return (const float*)d_in[i]; };

    float* Wk = (float*)d_ws;
    size_t off = 0;
    auto alloc = [&](size_t n) -> float* {
        float* p = Wk + off;
        off += (n + 63) & ~(size_t)63;
        return p;
    };
    float* b_up = alloc(1048576);   // conv-transpose output (C, P)
    float* b_t  = alloc(1048576);   // (L, C) tokens
    float* b_xz = alloc(4194304);   // (L, 2*Di)
    float* b_xc = alloc(2097152);   // (L, Di)
    float* b_dbl= alloc(1114112);   // (L, Nx)
    float* b_dt = alloc(2097152);   // (L, Di)
    float* b_ys = alloc(2097152);   // (L, Di)
    float* b_g  = alloc(2097152);   // (L, Di)
    float* b_mo = alloc(1048576);   // (L, C) mamba out
    float* b_h1 = alloc(4194304);   // (L, 4C)
    float* b_so = alloc(1048576);   // (L, C) stage out
    float* b_cA = alloc(65536);
    float* b_cH = alloc(65536);
    float* b_cI = alloc(65536);

    const float* skip_in[2] = { P(1), P(0) };   // stage0: feat1, stage1: feat0

    for (int st = 0; st < 2; ++st) {
        int base = 3 + st * 21;
        const float *up_w = P(base + 0), *up_b = P(base + 1), *in_g = P(base + 2),
                    *in_b = P(base + 3), *fuse_w = P(base + 4), *fuse_b = P(base + 5),
                    *ln_g = P(base + 6), *ln_b = P(base + 7), *in_proj = P(base + 8),
                    *conv_w = P(base + 9), *conv_b = P(base + 10), *x_proj = P(base + 11),
                    *dt_w = P(base + 12), *dt_b = P(base + 13), *A_log = P(base + 14),
                    *Dp = P(base + 15), *out_proj = P(base + 16), *mlp_w1 = P(base + 17),
                    *mlp_b1 = P(base + 18), *mlp_w2 = P(base + 19), *mlp_b2 = P(base + 20);

        const int Cin = (st == 0) ? 64 : 32;
        const int C   = Cin / 2;
        const int Hin = (st == 0) ? 64 : 128;
        const int Hout = 2 * Hin;
        const int L = Hout * Hout;
        const int Di = 2 * C;
        const int R = (C + 15) / 16;
        const int Nx = R + 2 * DSTATE;
        const int CL = 256;
        const int nch = L / CL;

        const float* ct_in = (st == 0) ? P(2) : b_so;
        const int in_lc = (st == 0) ? 0 : 1;

        { int tot = C * L;
          vmd_convt4x4<<<(tot + 255) / 256, 256, 0, stream>>>(ct_in, in_lc, up_w, up_b,
                                                              b_up, Cin, C, Hin, Hin); }
        vmd_inorm_lrelu<<<C, 256, 0, stream>>>(b_up, in_g, in_b, L);
        { int tot = L * C;
          vmd_fuse1x1<<<(tot + 255) / 256, 256, 0, stream>>>(b_up, skip_in[st], fuse_w,
                                                             fuse_b, b_t, C, L); }
        vmd_layernorm<<<(L + 7) / 8, 256, 0, stream>>>(b_t, ln_g, ln_b, L, C);

        launch_gemm(b_t, in_proj, nullptr, b_xz, L, C, 2 * Di, 0, stream);

        { int tot = L * Di;
          vmd_dwconv_silu<<<(tot + 255) / 256, 256, 0, stream>>>(b_xz, conv_w, conv_b,
                                                                 b_xc, L, Di); }
        { int tot = L * Nx;
          vmd_xproj<<<(tot + 255) / 256, 256, 0, stream>>>(b_xc, x_proj, b_dbl, L, Di, Nx); }
        { int tot = L * Di;
          vmd_dtcalc<<<(tot + 255) / 256, 256, 0, stream>>>(b_dbl, dt_w, dt_b, b_dt,
                                                            L, Di, R, Nx); }
        { int tot = Di * DSTATE * nch;
          vmd_scan_p1<<<(tot + 255) / 256, 256, 0, stream>>>(b_dt, b_dbl, b_xc, A_log,
                                                             b_cA, b_cH, Di, Nx, R, nch, CL); }
        { int tot = Di * DSTATE;
          vmd_scan_p2<<<(tot + 255) / 256, 256, 0, stream>>>(b_cA, b_cH, b_cI, Di, nch); }
        { int tot = Di * DSTATE * nch;
          vmd_scan_p3<<<(tot + 255) / 256, 256, 0, stream>>>(b_dt, b_dbl, b_xc, A_log,
                                                             b_cI, b_ys, Di, Nx, R, nch, CL); }
        { int tot = L * Di;
          vmd_gate<<<(tot + 255) / 256, 256, 0, stream>>>(b_ys, b_xc, b_xz, Dp, b_g, L, Di); }

        launch_gemm(b_g, out_proj, nullptr, b_mo, L, Di, C, 0, stream);
        launch_gemm(b_mo, mlp_w1, mlp_b1, b_h1, L, C, 4 * C, 1, stream);   // gelu
        launch_gemm(b_h1, mlp_w2, mlp_b2, b_so, L, 4 * C, C, 0, stream);
    }

    vmd_outconv<<<(256 * 256 + 255) / 256, 256, 0, stream>>>(b_so, P(45), P(46),
                                                             (float*)d_out, 16, 256, 256);
}
